// SOC_LM_22230750724313
// MI455X (gfx1250) — matrix operand order repaired and checked by reading, not compiled, not run on hardware
//
#include <hip/hip_runtime.h>

// ---------------- model constants ----------------
enum {
  KB = 4, KT = 1024, NTOK = KB * KT,        // 4096 tokens
  D_MODEL = 768, D_INNER = 1536,
  LDXZ = 2 * D_INNER,                        // 3072
  D_STATE = 16, DT_RANK = 48, D_CONV = 4,
  LDDBC = DT_RANK + 2 * D_STATE,             // 80
  N_LAYERS = 4, N_FEAT = 12
};

typedef __bf16 bf16_t;
typedef __attribute__((ext_vector_type(16))) __bf16 v16bf;
typedef __attribute__((ext_vector_type(8)))  float  v8f;

__device__ inline v16bf pack2(uint4 a, uint4 b) {
  union { uint4 u[2]; v16bf v; } x;
  x.u[0] = a; x.u[1] = b;
  return x.v;
}

__device__ inline float silu_f(float x) { return x / (1.f + __expf(-x)); }

// LDS byte address = low 32 bits of the generic (flat-aperture) pointer (ISA 10.2).
__device__ inline unsigned lds_lo32(const void* p) {
  return (unsigned)(unsigned long long)p;
}

// CDNA5 async copy global -> LDS, 16B per lane, tracked by ASYNCcnt (ISA 15.18.3 op 98).
__device__ inline void async_ld_b128(unsigned lds_off, const void* gptr) {
  asm volatile("global_load_async_to_lds_b128 %0, %1, off"
               :: "v"(lds_off), "v"(gptr)
               : "memory");
}
#define WAIT_ASYNCCNT(n) asm volatile("s_wait_asynccnt " #n ::: "memory")

// ---------------- f32 -> bf16 conversion ----------------
__global__ void k_cvt_bf16(const float* __restrict__ src, bf16_t* __restrict__ dst, int n) {
  int i = blockIdx.x * blockDim.x + threadIdx.x;
  if (i < n) dst[i] = (bf16_t)src[i];
}

// ---------------- input projection: h = x @ inp_w^T + b ----------------
__global__ void k_input_proj(const float* __restrict__ x, const float* __restrict__ w,
                             const float* __restrict__ b, float* __restrict__ h) {
  int i = blockIdx.x * blockDim.x + threadIdx.x;
  if (i >= NTOK * D_MODEL) return;
  int d = i % D_MODEL;
  int token = i / D_MODEL;
  const float* xr = x + (size_t)token * N_FEAT;
  const float* wr = w + (size_t)d * N_FEAT;
  float acc = b[d];
#pragma unroll
  for (int f = 0; f < N_FEAT; ++f) acc += xr[f] * wr[f];
  h[i] = acc;
}

// ---------------- rmsnorm -> bf16 ----------------
__global__ __launch_bounds__(256) void k_rmsnorm_bf16(const float* __restrict__ h,
                                                      const float* __restrict__ w,
                                                      bf16_t* __restrict__ out, int D) {
  int token = blockIdx.x, tid = threadIdx.x;
  const float* row = h + (size_t)token * D;
  __shared__ float red[256];
  float ss = 0.f;
  for (int i = tid; i < D; i += 256) { float v = row[i]; ss += v * v; }
  red[tid] = ss; __syncthreads();
  for (int s = 128; s > 0; s >>= 1) { if (tid < s) red[tid] += red[tid + s]; __syncthreads(); }
  float scale = rsqrtf(red[0] / (float)D + 1e-5f);
  for (int i = tid; i < D; i += 256)
    out[(size_t)token * D + i] = (bf16_t)(row[i] * scale * w[i]);
}

// ---------------- WMMA bf16 GEMM:  C[M,N] = A[M,K] * W[N,K]^T (+residual) ----------------
// Block: 256 threads = 8 waves (2 Mwaves x 4 Nwaves). Block tile 64x128, wave tile 32x32,
// K staged 32 at a time through double-buffered LDS filled by GLOBAL_LOAD_ASYNC_TO_LDS_B128.
// Fragment packing: A and B both use contiguous K order (k = 16*half + e).
__global__ __launch_bounds__(256) void k_gemm_bf16(const bf16_t* __restrict__ A,
                                                   const bf16_t* __restrict__ Bw,
                                                   float* __restrict__ C,
                                                   const float* __restrict__ residual,
                                                   int M, int N, int K) {
  __shared__ __attribute__((aligned(16))) bf16_t As[2][64][40];
  __shared__ __attribute__((aligned(16))) bf16_t Bs[2][128][40];

  const int tid  = threadIdx.x;
  const int lane = tid & 31;
  const int wave = tid >> 5;
  const int wm   = wave & 1;     // 0..1 : rows 32*wm
  const int wn   = wave >> 1;    // 0..3 : cols 32*wn
  const int m0   = blockIdx.y * 64;
  const int n0   = blockIdx.x * 128;

  // staging: A tile 64x32 (1 x b128/thread), B tile 128x32 (2 x b128/thread)
  const int lrow = tid >> 2;        // 0..63
  const int lcol = (tid & 3) * 8;   // 0,8,16,24
  const int arow = m0 + lrow;       // M is a multiple of 64 in this model
  int brow0 = n0 + lrow;       if (brow0 >= N) brow0 = N - 1;  // clamp; stores guarded
  int brow1 = n0 + 64 + lrow;  if (brow1 >= N) brow1 = N - 1;

  const bf16_t* Ap  = A  + (size_t)arow  * K + lcol;
  const bf16_t* Bp0 = Bw + (size_t)brow0 * K + lcol;
  const bf16_t* Bp1 = Bw + (size_t)brow1 * K + lcol;

  unsigned aoff[2], boff0[2], boff1[2];
#pragma unroll
  for (int s = 0; s < 2; ++s) {
    aoff[s]  = lds_lo32(&As[s][lrow][lcol]);
    boff0[s] = lds_lo32(&Bs[s][lrow][lcol]);
    boff1[s] = lds_lo32(&Bs[s][64 + lrow][lcol]);
  }

  const int half = lane >> 4;       // 0/1 lane half
  const int l16  = lane & 15;
  const int akb  = half * 16;       // A-fragment K base (contiguous K, matches B)
  const int bkb  = half * 16;       // B-fragment K base (contiguous K, matches A)

  v8f acc00 = {}, acc01 = {}, acc10 = {}, acc11 = {};

  // stage 0 in flight
  async_ld_b128(aoff[0],  Ap);
  async_ld_b128(boff0[0], Bp0);
  async_ld_b128(boff1[0], Bp1);

  const int nsteps = K / 32;
  for (int s = 0; s < nsteps; ++s) {
    const int cur = s & 1;
    const int k0  = s * 32;

    __syncthreads();   // everyone done reading buffer cur^1 before refilling it
    if (k0 + 32 < K) {                       // uniform branch
      async_ld_b128(aoff[cur ^ 1],  Ap  + k0 + 32);
      async_ld_b128(boff0[cur ^ 1], Bp0 + k0 + 32);
      async_ld_b128(boff1[cur ^ 1], Bp1 + k0 + 32);
      WAIT_ASYNCCNT(0x3);                    // leave only next-stage copies in flight
      if (k0 + 64 < K) {                     // prefetch two stages ahead
        __builtin_prefetch(Ap  + k0 + 64, 0, 3);
        __builtin_prefetch(Bp0 + k0 + 64, 0, 3);
        __builtin_prefetch(Bp1 + k0 + 64, 0, 3);
      }
    } else {
      WAIT_ASYNCCNT(0x0);
    }
    __syncthreads();   // all waves' stage-cur data visible in LDS

    // B fragments: column n = 32*wn + {0,16} + l16, K = bkb + e
    v16bf bf0, bf1;
    {
      const bf16_t* p = &Bs[cur][wn * 32 + l16][bkb];
      bf0 = pack2(*(const uint4*)p, *(const uint4*)(p + 8));
      p = &Bs[cur][wn * 32 + 16 + l16][bkb];
      bf1 = pack2(*(const uint4*)p, *(const uint4*)(p + 8));
    }
    // A fragments: rows (wm*32 + {0,16} + l16), K = akb + e
    v16bf af0, af1;
    {
      const bf16_t* p = &As[cur][wm * 32 + l16][akb];
      af0 = pack2(*(const uint4*)p, *(const uint4*)(p + 8));
      p = &As[cur][wm * 32 + 16 + l16][akb];
      af1 = pack2(*(const uint4*)p, *(const uint4*)(p + 8));
    }

    acc00 = __builtin_amdgcn_wmma_f32_16x16x32_bf16(false, af0, false, bf0, (short)0, acc00, false, false);
    acc01 = __builtin_amdgcn_wmma_f32_16x16x32_bf16(false, af0, false, bf1, (short)0, acc01, false, false);
    acc10 = __builtin_amdgcn_wmma_f32_16x16x32_bf16(false, af1, false, bf0, (short)0, acc10, false, false);
    acc11 = __builtin_amdgcn_wmma_f32_16x16x32_bf16(false, af1, false, bf1, (short)0, acc11, false, false);
  }

  // Epilogue: C/D layout — element v: row = v + half*8, col = l16 (ISA 32-bit C/D layout)
  const int c0 = n0 + wn * 32 + l16;
  const int c1 = c0 + 16;
  auto store_tile = [&](const v8f& acc, int mbase, int col) {
    if (col < N) {
#pragma unroll
      for (int v = 0; v < 8; ++v) {
        int row = mbase + half * 8 + v;
        size_t idx = (size_t)row * N + col;
        float val = acc[v];
        C[idx] = residual ? residual[idx] + val : val;
      }
    }
  };
  store_tile(acc00, m0 + wm * 32,      c0);
  store_tile(acc01, m0 + wm * 32,      c1);
  store_tile(acc10, m0 + wm * 32 + 16, c0);
  store_tile(acc11, m0 + wm * 32 + 16, c1);
}

// ---------------- depthwise causal conv (4 taps) + bias + SiLU ----------------
__global__ void k_conv_silu(const float* __restrict__ xz, const float* __restrict__ cw,
                            const float* __restrict__ cb, float* __restrict__ xcf,
                            bf16_t* __restrict__ xcb) {
  int i = blockIdx.x * blockDim.x + threadIdx.x;
  if (i >= NTOK * D_INNER) return;
  int d = i % D_INNER;
  int token = i / D_INNER;
  int t = token % KT;
  const float* w4 = cw + (size_t)d * D_CONV;
  float acc = cb[d];
#pragma unroll
  for (int j = 0; j < D_CONV; ++j) {
    int tt = t - (D_CONV - 1) + j;
    if (tt >= 0) acc += w4[j] * xz[(size_t)(token - (D_CONV - 1) + j) * LDXZ + d];
  }
  float s = silu_f(acc);
  xcf[i] = s;
  xcb[i] = (bf16_t)s;
}

// ---------------- dt_proj + softplus ----------------
__global__ void k_dt(const float* __restrict__ dbc, const float* __restrict__ dtw,
                     const float* __restrict__ dtb, float* __restrict__ delta) {
  int i = blockIdx.x * blockDim.x + threadIdx.x;
  if (i >= NTOK * D_INNER) return;
  int d = i % D_INNER;
  int token = i / D_INNER;
  const float* r = dbc + (size_t)token * LDDBC;
  const float* w = dtw + (size_t)d * DT_RANK;
  float acc = dtb[d];
#pragma unroll 8
  for (int k = 0; k < DT_RANK; ++k) acc += r[k] * w[k];
  delta[i] = (acc > 20.f) ? acc : log1pf(__expf(acc));
}

// ---------------- selective scan (fused D-skip + SiLU gate, bf16 output) ----------------
__global__ __launch_bounds__(256) void k_scan(const float* __restrict__ delta,
                                              const float* __restrict__ xcf,
                                              const float* __restrict__ xz,
                                              const float* __restrict__ dbc,
                                              const float* __restrict__ A_log,
                                              const float* __restrict__ Dsk,
                                              bf16_t* __restrict__ ybf) {
  int b = blockIdx.y;
  int d = blockIdx.x * blockDim.x + threadIdx.x;   // channel
  float a[D_STATE], h[D_STATE];
#pragma unroll
  for (int n = 0; n < D_STATE; ++n) {
    a[n] = -__expf(A_log[(size_t)d * D_STATE + n]);
    h[n] = 0.f;
  }
  float dsk = Dsk[d];
  __shared__ float sB[D_STATE], sC[D_STATE];

  for (int t = 0; t < KT; ++t) {
    size_t token = (size_t)b * KT + t;
    __syncthreads();
    if (threadIdx.x < 2 * D_STATE) {
      float v = dbc[token * LDDBC + DT_RANK + threadIdx.x];
      if (threadIdx.x < D_STATE) sB[threadIdx.x] = v;
      else                       sC[threadIdx.x - D_STATE] = v;
    }
    __syncthreads();
    float dt = delta[token * D_INNER + d];
    float xv = xcf[token * D_INNER + d];
    float dx = dt * xv;
    float y = 0.f;
#pragma unroll
    for (int n = 0; n < D_STATE; ++n) {
      h[n] = __expf(dt * a[n]) * h[n] + dx * sB[n];
      y += h[n] * sC[n];
    }
    float z = xz[token * LDXZ + D_INNER + d];
    ybf[token * D_INNER + d] = (bf16_t)((y + dsk * xv) * silu_f(z));
  }
}

// ---------------- final rmsnorm + head ----------------
__global__ __launch_bounds__(256) void k_final_head(const float* __restrict__ h,
                                                    const float* __restrict__ norm_w,
                                                    const float* __restrict__ head_w,
                                                    const float* __restrict__ head_b,
                                                    float* __restrict__ out) {
  int token = blockIdx.x, tid = threadIdx.x;
  const float* row = h + (size_t)token * D_MODEL;
  __shared__ float red[256];
  float ss = 0.f;
  for (int i = tid; i < D_MODEL; i += 256) { float v = row[i]; ss += v * v; }
  red[tid] = ss; __syncthreads();
  for (int s = 128; s > 0; s >>= 1) { if (tid < s) red[tid] += red[tid + s]; __syncthreads(); }
  float scale = rsqrtf(red[0] / (float)D_MODEL + 1e-5f);
  __syncthreads();
  float dot = 0.f;
  for (int i = tid; i < D_MODEL; i += 256) dot += row[i] * scale * norm_w[i] * head_w[i];
  red[tid] = dot; __syncthreads();
  for (int s = 128; s > 0; s >>= 1) { if (tid < s) red[tid] += red[tid + s]; __syncthreads(); }
  if (tid == 0) out[token] = red[0] + head_b[0];
}

// ---------------- host side ----------------
static inline size_t align256(size_t x) { return (x + 255) & ~(size_t)255; }

extern "C" void kernel_launch(void* const* d_in, const int* in_sizes, int n_in,
                              void* d_out, int out_size, void* d_ws, size_t ws_size,
                              hipStream_t stream) {
  const float* x         = (const float*)d_in[0];
  const float* inp_w     = (const float*)d_in[1];
  const float* inp_b     = (const float*)d_in[2];
  const float* in_proj_w = (const float*)d_in[3];
  const float* conv_w    = (const float*)d_in[4];
  const float* conv_b    = (const float*)d_in[5];
  const float* x_proj_w  = (const float*)d_in[6];
  const float* dt_proj_w = (const float*)d_in[7];
  const float* dt_proj_b = (const float*)d_in[8];
  const float* A_log     = (const float*)d_in[9];
  const float* D_skip    = (const float*)d_in[10];
  const float* out_proj_w= (const float*)d_in[11];
  const float* ln_w      = (const float*)d_in[12];
  const float* norm_w    = (const float*)d_in[13];
  const float* head_w    = (const float*)d_in[14];
  const float* head_b    = (const float*)d_in[15];
  float* out = (float*)d_out;

  char* ws = (char*)d_ws;
  auto grab = [&](size_t bytes) -> char* { char* p = ws; ws += align256(bytes); return p; };

  float*  h     = (float*) grab((size_t)NTOK * D_MODEL * 4);
  bf16_t* xn    = (bf16_t*)grab((size_t)NTOK * D_MODEL * 2);
  float*  xz    = (float*) grab((size_t)NTOK * LDXZ * 4);
  float*  xcf   = (float*) grab((size_t)NTOK * D_INNER * 4);
  bf16_t* xcb   = (bf16_t*)grab((size_t)NTOK * D_INNER * 2);
  float*  dbc   = (float*) grab((size_t)NTOK * LDDBC * 4);
  float*  delta = (float*) grab((size_t)NTOK * D_INNER * 4);
  bf16_t* ybf   = (bf16_t*)grab((size_t)NTOK * D_INNER * 2);
  bf16_t* w_in  = (bf16_t*)grab((size_t)N_LAYERS * LDXZ * D_MODEL * 2);
  bf16_t* w_xp  = (bf16_t*)grab((size_t)N_LAYERS * LDDBC * D_INNER * 2);
  bf16_t* w_out = (bf16_t*)grab((size_t)N_LAYERS * D_MODEL * D_INNER * 2);

  // weight conversions (deterministic, every call)
  {
    int n1 = N_LAYERS * LDXZ * D_MODEL;
    int n2 = N_LAYERS * LDDBC * D_INNER;
    int n3 = N_LAYERS * D_MODEL * D_INNER;
    k_cvt_bf16<<<(n1 + 255) / 256, 256, 0, stream>>>(in_proj_w, w_in, n1);
    k_cvt_bf16<<<(n2 + 255) / 256, 256, 0, stream>>>(x_proj_w, w_xp, n2);
    k_cvt_bf16<<<(n3 + 255) / 256, 256, 0, stream>>>(out_proj_w, w_out, n3);
  }

  k_input_proj<<<(NTOK * D_MODEL + 255) / 256, 256, 0, stream>>>(x, inp_w, inp_b, h);

  for (int l = 0; l < N_LAYERS; ++l) {
    k_rmsnorm_bf16<<<NTOK, 256, 0, stream>>>(h, ln_w + (size_t)l * D_MODEL, xn, D_MODEL);

    // xz = xn @ in_proj_w^T   [4096 x 3072]
    k_gemm_bf16<<<dim3(LDXZ / 128, NTOK / 64), 256, 0, stream>>>(
        xn, w_in + (size_t)l * LDXZ * D_MODEL, xz, nullptr, NTOK, LDXZ, D_MODEL);

    k_conv_silu<<<(NTOK * D_INNER + 255) / 256, 256, 0, stream>>>(
        xz, conv_w + (size_t)l * D_INNER * D_CONV, conv_b + (size_t)l * D_INNER, xcf, xcb);

    // dbc = xc @ x_proj_w^T   [4096 x 80]
    k_gemm_bf16<<<dim3((LDDBC + 127) / 128, NTOK / 64), 256, 0, stream>>>(
        xcb, w_xp + (size_t)l * LDDBC * D_INNER, dbc, nullptr, NTOK, LDDBC, D_INNER);

    k_dt<<<(NTOK * D_INNER + 255) / 256, 256, 0, stream>>>(
        dbc, dt_proj_w + (size_t)l * D_INNER * DT_RANK, dt_proj_b + (size_t)l * D_INNER, delta);

    k_scan<<<dim3(D_INNER / 256, KB), 256, 0, stream>>>(
        delta, xcf, xz, dbc, A_log + (size_t)l * D_INNER * D_STATE,
        D_skip + (size_t)l * D_INNER, ybf);

    // h = h + y @ out_proj_w^T   [4096 x 768]
    k_gemm_bf16<<<dim3(D_MODEL / 128, NTOK / 64), 256, 0, stream>>>(
        ybf, w_out + (size_t)l * D_MODEL * D_INNER, h, h, NTOK, D_MODEL, D_INNER);
  }

  k_final_head<<<NTOK, 256, 0, stream>>>(h, norm_w, head_w, head_b, out);
}
